// TreeLSTMCell_36447092473860
// MI455X (gfx1250) — matrix operand order repaired and checked by reading, not compiled, not run on hardware
//
#include <hip/hip_runtime.h>

typedef __bf16 bf16_t;
typedef __bf16 v16bf __attribute__((ext_vector_type(16)));
typedef __bf16 v8bf  __attribute__((ext_vector_type(8)));
typedef float  v8f   __attribute__((ext_vector_type(8)));

#define HDIM 128
#define NW   4     // waves per block
#define MT   16    // rows (nodes) per wave tile

// LDS strides (in elements) — padded so 16 rows hit distinct banks
#define S_HIOU 264 // bf16, 256 cols used
#define S_EMB  136 // bf16, 128 cols used
#define S_CT   136 // bf16
#define S_F32  132 // f32,  128 cols used

// per-wave LDS byte offsets (all 16B aligned)
#define OF_HIOU 0       // bf16 [16][264]     8448B
#define OF_EMB  8448    // bf16 [16][136]     4352B
#define OF_CT   12800   // bf16 [2][16][136]  8704B  (dead after forget phase)
#define OF_BI   12800   // f32  [16][132]     aliases CT (i_ then c)
#define OF_SCR  21504   // f32  [16][132]     8448B
#define OF_CC   29952   // f32  [16][132]     8448B
#define WBYTES  38400

__device__ __forceinline__ v16bf ld_a(const bf16_t* base, int stride, int kt, int lane) {
  // 16-bit A 16x32 layout: lane&15 = row M; hi half-wave holds K 8..15 / 24..31
  const bf16_t* p = base + (lane & 15) * stride + kt * 32 + (lane >> 4) * 8;
  v8bf lo = *(const v8bf*)p;        // K chunk 0..7 (or 8..15)
  v8bf hi = *(const v8bf*)(p + 16); // K chunk +16
  return __builtin_shufflevector(lo, hi, 0,1,2,3,4,5,6,7,8,9,10,11,12,13,14,15);
}

// B fragment load: lanebase already carries per-lane (col,khalf) offsets;
// elemoff is a wave-uniform element offset (colbase*K + kt*32).
__device__ __forceinline__ v16bf ld_b2(const bf16_t* lanebase, int elemoff) {
  const bf16_t* p = lanebase + elemoff;
  v8bf lo = *(const v8bf*)p;
  v8bf hi = *(const v8bf*)(p + 16);
  return __builtin_shufflevector(lo, hi, 0,1,2,3,4,5,6,7,8,9,10,11,12,13,14,15);
}

__device__ __forceinline__ void st_c(float* buf, int nbase, const v8f& acc, int lane) {
  // f32 C 16x16 layout: lanes 0-15 rows 0-7, lanes 16-31 rows 8-15
  float* p = buf + (lane >> 4) * 8 * S_F32 + nbase + (lane & 15);
#pragma unroll
  for (int r = 0; r < 8; ++r) p[r * S_F32] = acc[r];
}

// out_scr[16][132] = A1[16,K1]@B1[coloff1:+128,:K1]^T + A2[16,K2]@B2[coloff2:+128,:K2]^T
// A fragments hoisted to registers; B loads software-pipelined (depth PF).
template <int NK1, int NK2, int PF>
__device__ __forceinline__ void gemm_chunk(float* out_scr,
    const bf16_t* A1, int lda1, const bf16_t* B1, int K1, int coloff1,
    const bf16_t* A2, int lda2, const bf16_t* B2, int K2, int coloff2,
    int lane) {
  constexpr int TOT = NK1 + NK2;
  v16bf afr[TOT];
#pragma unroll
  for (int kt = 0; kt < NK1; ++kt) afr[kt]       = ld_a(A1, lda1, kt, lane);
#pragma unroll
  for (int kt = 0; kt < NK2; ++kt) afr[NK1 + kt] = ld_a(A2, lda2, kt, lane);

  // per-lane base pointers (vector part of the address, computed once)
  const bf16_t* b1base = B1 + (size_t)(lane & 15) * (size_t)K1 + (lane >> 4) * 8;
  const bf16_t* b2base = B2 + (size_t)(lane & 15) * (size_t)K2 + (lane >> 4) * 8;

  for (int np = 0; np < 8; np += 2) {
    const int o1a = (coloff1 + np * 16) * K1;       // uniform element offsets
    const int o1b = o1a + 16 * K1;
    const int o2a = (coloff2 + np * 16) * K2;
    const int o2b = o2a + 16 * K2;

    v8f acc0 = {}; v8f acc1 = {};
    v16bf bb0[PF], bb1[PF];

    auto ldc = [&](int kt, int which) -> v16bf {
      if (kt < NK1) return ld_b2(b1base, (which ? o1b : o1a) + kt * 32);
      return ld_b2(b2base, (which ? o2b : o2a) + (kt - NK1) * 32);
    };

#pragma unroll
    for (int i = 0; i < PF - 1; ++i) { bb0[i] = ldc(i, 0); bb1[i] = ldc(i, 1); }

#pragma unroll
    for (int kt = 0; kt < TOT; ++kt) {
      const int nx = kt + PF - 1;
      if (nx < TOT) { bb0[nx % PF] = ldc(nx, 0); bb1[nx % PF] = ldc(nx, 1); }
      acc0 = __builtin_amdgcn_wmma_f32_16x16x32_bf16(false, afr[kt], false, bb0[kt % PF], (short)0, acc0, false, false);
      acc1 = __builtin_amdgcn_wmma_f32_16x16x32_bf16(false, afr[kt], false, bb1[kt % PF], (short)0, acc1, false, false);
    }
    st_c(out_scr, np * 16,      acc0, lane);
    st_c(out_scr, np * 16 + 16, acc1, lane);
  }
}

__device__ __forceinline__ void ln_stats(const float* buf, const float* bias, int lane,
                                         float& mu, float& inv) {
  int row = lane & 15, half = lane >> 4;
  const float* p = buf + row * S_F32 + half * 64;
  float s = 0.f, s2 = 0.f;
  if (bias) {
    const float* b = bias + half * 64;
    for (int j = 0; j < 64; ++j) { float x = p[j] + b[j]; s += x; s2 += x * x; }
  } else {
    for (int j = 0; j < 64; ++j) { float x = p[j]; s += x; s2 += x * x; }
  }
  s  += __shfl_xor(s, 16);
  s2 += __shfl_xor(s2, 16);
  mu  = s * (1.0f / 128.0f);
  inv = rsqrtf(s2 * (1.0f / 128.0f) - mu * mu + 1e-5f);
}

__global__ void cvt_f32_to_bf16(const float* __restrict__ src, bf16_t* __restrict__ dst, int n) {
  int i = blockIdx.x * blockDim.x + threadIdx.x;
  if (i < n) dst[i] = (bf16_t)src[i];
}

__global__ __launch_bounds__(NW * 32, 1)
void treelstm_fused(const float* __restrict__ emb, const float* __restrict__ h_child,
                    const float* __restrict__ c_child, const int* __restrict__ n_type,
                    const bf16_t* __restrict__ Wf, const bf16_t* __restrict__ Uf,
                    const bf16_t* __restrict__ Wiou, const bf16_t* __restrict__ Uiou,
                    const float* __restrict__ lin_biou, const float* __restrict__ lin_bf,
                    const float* __restrict__ gi, const float* __restrict__ bi,
                    const float* __restrict__ go, const float* __restrict__ bo,
                    const float* __restrict__ gu, const float* __restrict__ bu,
                    const float* __restrict__ gnf, const float* __restrict__ bnf,
                    const float* __restrict__ gnc, const float* __restrict__ bnc,
                    float* __restrict__ out, int Ntot) {
  __shared__ __align__(16) char smem[NW * WBYTES];
  const int lane = threadIdx.x & 31;
  const int wid  = threadIdx.x >> 5;
  char* wb = smem + wid * WBYTES;
  bf16_t* sHIOU = (bf16_t*)(wb + OF_HIOU);
  bf16_t* sEMB  = (bf16_t*)(wb + OF_EMB);
  bf16_t* sCT   = (bf16_t*)(wb + OF_CT);
  float*  sBI   = (float*)(wb + OF_BI);
  float*  sSCR  = (float*)(wb + OF_SCR);
  float*  sCC   = (float*)(wb + OF_CC);

  const long long node0 = ((long long)blockIdx.x * NW + wid) * MT;
  if (node0 + MT > Ntot) return;
  const size_t NH = (size_t)Ntot * HDIM;

  // ---- Phase 1: typed child aggregation (memory bound) ----
  for (int e = lane; e < MT * HDIM; e += 32) {
    int r = e >> 7, h = e & 127;
    long long n = node0 + r;
    float ht0 = 0.f, ht1 = 0.f, ct0 = 0.f, ct1 = 0.f;
#pragma unroll
    for (int k = 0; k < 4; ++k) {
      int t = n_type[n * 4 + k];
      float hv = h_child[((n * 4 + k) << 7) + h];
      float cv = c_child[((n * 4 + k) << 7) + h];
      float m0 = (t == 0) ? 1.f : 0.f;
      float m1 = (t == 1) ? 1.f : 0.f;
      ht0 += hv * m0; ct0 += cv * m0;
      ht1 += hv * m1; ct1 += cv * m1;
    }
    sHIOU[r * S_HIOU + h]       = (bf16_t)ht0;
    sHIOU[r * S_HIOU + 128 + h] = (bf16_t)ht1;
    sCT[r * S_CT + h]               = (bf16_t)ct0;
    sCT[MT * S_CT + r * S_CT + h]   = (bf16_t)ct1;
    sEMB[r * S_EMB + h] = (bf16_t)emb[(n << 7) + h];
  }
  // waves are independent; same-wave LDS ops are in-order (no barrier needed)

  const int row = lane & 15, half = lane >> 4;
  const int cb = half * 64;
  const long long nrow = node0 + row;

  // ---- Phase 2: forget gates, c_cell = sum_t sigmoid(LN(X + f_t + b_f)) * ct_t ----
  for (int t = 0; t < 2; ++t) {
    gemm_chunk<4, 8, 4>(sSCR, sEMB, S_EMB, Wf, 128, 0,
                        sHIOU, S_HIOU, Uf, 256, t * 128, lane);
    float mu, inv;
    ln_stats(sSCR, lin_bf, lane, mu, inv);
    const bf16_t* ct = sCT + t * MT * S_CT + row * S_CT + cb;
    const float*  sp = sSCR + row * S_F32 + cb;
    float*        cc = sCC  + row * S_F32 + cb;
    for (int j = 0; j < 64; ++j) {
      int c = cb + j;
      float y = (sp[j] + lin_bf[c] - mu) * inv * gnf[c] + bnf[c];
      float F = 1.f / (1.f + __expf(-y));
      float v = F * (float)ct[j];
      if (t == 0) cc[j] = v; else cc[j] += v;
    }
  }

  // ---- Phase 3: i_ = sigmoid(LN(iou[:,0:128])) -> sBI (CT now dead) ----
  {
    gemm_chunk<4, 8, 4>(sSCR, sEMB, S_EMB, Wiou, 128, 0,
                        sHIOU, S_HIOU, Uiou, 256, 0, lane);
    float mu, inv;
    ln_stats(sSCR, lin_biou, lane, mu, inv);
    const float* sp = sSCR + row * S_F32 + cb;
    float*       bp = sBI  + row * S_F32 + cb;
    for (int j = 0; j < 64; ++j) {
      int c = cb + j;
      float y = (sp[j] + lin_biou[c] - mu) * inv * gi[c] + bi[c];
      bp[j] = 1.f / (1.f + __expf(-y));
    }
  }

  // ---- Phase 4: u_ = tanh(LN(iou[:,256:384])); c = i_*u_ + c_cell -> sBI, store c ----
  {
    gemm_chunk<4, 8, 4>(sSCR, sEMB, S_EMB, Wiou, 128, 256,
                        sHIOU, S_HIOU, Uiou, 256, 256, lane);
    float mu, inv;
    ln_stats(sSCR, lin_biou + 256, lane, mu, inv);
    const float* sp = sSCR + row * S_F32 + cb;
    float*       bp = sBI  + row * S_F32 + cb;
    const float* cc = sCC  + row * S_F32 + cb;
    for (int j = 0; j < 64; ++j) {
      int c = cb + j;
      float y  = (sp[j] + lin_biou[256 + c] - mu) * inv * gu[c] + bu[c];
      float u_ = tanhf(y);
      float cv = bp[j] * u_ + cc[j];
      bp[j] = cv;                                   // sBI now holds c
      out[NH + (size_t)nrow * HDIM + c] = cv;       // out[1] = c
    }
  }

  // ---- Phase 5: o_ = sigmoid(LN(iou[:,128:256])) -> sSCR (in place) ----
  {
    gemm_chunk<4, 8, 4>(sSCR, sEMB, S_EMB, Wiou, 128, 128,
                        sHIOU, S_HIOU, Uiou, 256, 128, lane);
    float mu, inv;
    ln_stats(sSCR, lin_biou + 128, lane, mu, inv);
    float* sp = sSCR + row * S_F32 + cb;
    for (int j = 0; j < 64; ++j) {
      int c = cb + j;
      float y = (sp[j] + lin_biou[128 + c] - mu) * inv * go[c] + bo[c];
      sp[j] = 1.f / (1.f + __expf(-y));
    }
  }

  // ---- Phase 6: h = o_ * tanh(LN(c)) ----
  {
    float mu, inv;
    ln_stats(sBI, nullptr, lane, mu, inv);
    const float* cp = sBI  + row * S_F32 + cb;
    const float* op = sSCR + row * S_F32 + cb;
    for (int j = 0; j < 64; ++j) {
      int c = cb + j;
      float y = (cp[j] - mu) * inv * gnc[c] + bnc[c];
      out[(size_t)nrow * HDIM + c] = op[j] * tanhf(y);   // out[0] = h
    }
  }
}

extern "C" void kernel_launch(void* const* d_in, const int* in_sizes, int n_in,
                              void* d_out, int out_size, void* d_ws, size_t ws_size,
                              hipStream_t stream) {
  const float* emb      = (const float*)d_in[0];
  const float* h_child  = (const float*)d_in[1];
  const float* c_child  = (const float*)d_in[2];
  const int*   n_type   = (const int*)d_in[3];
  const float* W_iou    = (const float*)d_in[4];   // [384,128]
  const float* U_iou    = (const float*)d_in[5];   // [384,256]
  const float* b_iou    = (const float*)d_in[6];
  const float* W_f      = (const float*)d_in[7];   // [128,128]
  const float* U_f      = (const float*)d_in[8];   // [256,256]
  const float* b_f      = (const float*)d_in[9];
  const float* g_i = (const float*)d_in[10], *b_i = (const float*)d_in[11];
  const float* g_o = (const float*)d_in[12], *b_o = (const float*)d_in[13];
  const float* g_u = (const float*)d_in[14], *b_u = (const float*)d_in[15];
  const float* gn_f = (const float*)d_in[16], *bn_f = (const float*)d_in[17];
  const float* gn_c = (const float*)d_in[18], *bn_c = (const float*)d_in[19];

  bf16_t* ws = (bf16_t*)d_ws;
  bf16_t* wWf   = ws;            // 16384
  bf16_t* wUf   = ws + 16384;    // 65536
  bf16_t* wWiou = ws + 81920;    // 49152
  bf16_t* wUiou = ws + 131072;   // 98304

  cvt_f32_to_bf16<<<(16384 + 255) / 256, 256, 0, stream>>>(W_f,   wWf,   16384);
  cvt_f32_to_bf16<<<(65536 + 255) / 256, 256, 0, stream>>>(U_f,   wUf,   65536);
  cvt_f32_to_bf16<<<(49152 + 255) / 256, 256, 0, stream>>>(W_iou, wWiou, 49152);
  cvt_f32_to_bf16<<<(98304 + 255) / 256, 256, 0, stream>>>(U_iou, wUiou, 98304);

  int N = in_sizes[0] / HDIM;
  int nodes_per_block = NW * MT;  // 64
  dim3 grid((N + nodes_per_block - 1) / nodes_per_block);
  treelstm_fused<<<grid, NW * 32, 0, stream>>>(
      emb, h_child, c_child, n_type,
      wWf, wUf, wWiou, wUiou,
      b_iou, b_f, g_i, b_i, g_o, b_o, g_u, b_u, gn_f, bn_f, gn_c, bn_c,
      (float*)d_out, N);
}
